// AutoregressiveHead_22436909154878
// MI455X (gfx1250) — compile-verified, matrix operand order checked by reading, not run on hardware
//
#include <hip/hip_runtime.h>
#include <hip/hip_bf16.h>

// ---------------------------------------------------------------------------
// Autoregressive 2-layer GRU head for MI455X (gfx1250, wave32, WMMA).
//
//  * context mean kernel streams the 268MB encoded_features[:, -1] slice once
//    (HBM-bound: ~11.5us at 23.3 TB/s).
//  * weights pre-converted to f16 (1.7MB -> L2-resident); step_emb GEMM folded
//    into a precomputed per-step bias embbias[12][768].
//  * recurrence kernel: 64 WGs x 512 threads (16 waves) — each WG owns 16
//    batch rows for all 12 steps. v_wmma_f32_16x16x32_f16 with f32 accum,
//    h-state in LDS as f16. 16 waves/WG = 4 waves/SIMD32 for latency hiding
//    in the serial 12-step chain.
// ---------------------------------------------------------------------------

typedef _Float16 h16 __attribute__((ext_vector_type(16)));
typedef _Float16 h8  __attribute__((ext_vector_type(8)));
typedef float    f8  __attribute__((ext_vector_type(8)));

#define HSTR 264          // padded LDS row stride (halves) for 256-wide rows
#define WROWS 768         // gate rows per GRU layer (3*H)
#define HID 256
#define NTHREADS 512      // 16 waves per workgroup

__device__ __forceinline__ f8 wmma16x16x32(h16 a, h16 b, f8 c) {
    // v_wmma_f32_16x16x32_f16  (neg_a, A, neg_b, B, c_mod, C, reuse_a, reuse_b)
    return __builtin_amdgcn_wmma_f32_16x16x32_f16(false, a, false, b, (short)0, c,
                                                  false, false);
}

// A fragment (16x32, f16) from LDS, row-major with row stride HSTR halves.
// lane L: row = L%16, k = k0 + 16*(L/16), 16 consecutive halves (2x16B loads).
__device__ __forceinline__ h16 load_a_lds(const _Float16* base, int lane, int k0) {
    const int row = lane & 15;
    const int k   = k0 + ((lane >> 4) << 4);
    const h8* p = (const h8*)(base + row * HSTR + k);
    h8 lo = p[0];
    h8 hi = p[1];
    h16 r;
#pragma unroll
    for (int i = 0; i < 8; ++i) { r[i] = lo[i]; r[8 + i] = hi[i]; }
    return r;
}

// B fragment for X @ W^T: B[k][n] = W[n][k]; W is row-major (rows x 256) f16.
// lane L: n = n0 + L%16, k = k0 + 16*(L/16) -> one contiguous 32B load.
__device__ __forceinline__ h16 load_b_wt(const _Float16* __restrict__ W,
                                         int lane, int n0, int k0) {
    const int n = n0 + (lane & 15);
    const int k = k0 + ((lane >> 4) << 4);
    return *(const h16*)(W + (size_t)n * HID + k);
}

__device__ __forceinline__ float sigmoidf_(float x) {
    return 1.0f / (1.0f + __expf(-x));
}

// ---------------------------------------------------------------------------
// Prep kernel 1: convert weights f32 -> f16 into workspace.
// Layout (halves): Wp(768x256 = W_ih0[:,256:]) | Whh0 | Wih1 | Whh1 | Wout(256x256)
// ---------------------------------------------------------------------------
__global__ __launch_bounds__(256) void cvt_weights_kernel(
    const float* __restrict__ Wih0, const float* __restrict__ Whh0,
    const float* __restrict__ Wih1, const float* __restrict__ Whh1,
    const float* __restrict__ Wout, _Float16* __restrict__ wf16) {
    const int S = WROWS * HID;
    int idx = blockIdx.x * blockDim.x + threadIdx.x;
    if (idx < S) {
        int g = idx / HID, k = idx - g * HID;
        wf16[idx] = (_Float16)Wih0[(size_t)g * 512 + 256 + k];   // last_pred half
    } else if (idx < 2 * S) {
        wf16[idx] = (_Float16)Whh0[idx - S];
    } else if (idx < 3 * S) {
        wf16[idx] = (_Float16)Wih1[idx - 2 * S];
    } else if (idx < 4 * S) {
        wf16[idx] = (_Float16)Whh1[idx - 3 * S];
    } else if (idx < 4 * S + HID * HID) {
        wf16[idx] = (_Float16)Wout[idx - 4 * S];
    }
}

// ---------------------------------------------------------------------------
// Prep kernel 2: embbias[s][g] = step_emb[s] . W_ih0[g, :256] + b_ih0[g]
// ---------------------------------------------------------------------------
__global__ __launch_bounds__(256) void emb_bias_kernel(
    const float* __restrict__ step_emb, const float* __restrict__ Wih0,
    const float* __restrict__ b_ih0, float* __restrict__ eb) {
    int idx = blockIdx.x * blockDim.x + threadIdx.x;
    if (idx >= 12 * WROWS) return;
    int s = idx / WROWS, g = idx - s * WROWS;
    float acc = b_ih0[g];
    const float* se = step_emb + (size_t)s * 256;
    const float* w  = Wih0 + (size_t)g * 512;
#pragma unroll 8
    for (int k = 0; k < 256; ++k) acc += se[k] * w[k];
    eb[idx] = acc;
}

// ---------------------------------------------------------------------------
// Prep kernel 3: context[b][d] = mean_n encoded_features[b, T-1, n, d]  (f16)
// ---------------------------------------------------------------------------
__global__ __launch_bounds__(256) void context_kernel(
    const float* __restrict__ ef, _Float16* __restrict__ ctx) {
    const int b = blockIdx.x;
    const int d = threadIdx.x;
    const float* p = ef + (((size_t)b * 8 + 7) * 256) * 256 + d;
    float acc = 0.0f;
#pragma unroll 4
    for (int n = 0; n < 256; ++n) acc += p[(size_t)n * 256];
    ctx[(size_t)b * 256 + d] = (_Float16)(acc * (1.0f / 256.0f));
}

// ---------------------------------------------------------------------------
// Main recurrence kernel: one WG = 16 batch rows, 16 waves, 12 steps.
// ---------------------------------------------------------------------------
__global__ __launch_bounds__(NTHREADS) void ar_gru_kernel(
    const _Float16* __restrict__ wf16,
    const float* __restrict__ embbias,   // 12 x 768 (includes b_ih0 + emb GEMM)
    const _Float16* __restrict__ ctx,    // 1024 x 256
    const float* __restrict__ b_hh0,
    const float* __restrict__ b_ih1,
    const float* __restrict__ b_hh1,
    const float* __restrict__ b_out,
    float* __restrict__ out)             // (1024, 12, 256) f32
{
    __shared__ _Float16 hbuf[3][16 * HSTR];  // [0]=h0, [1]=h1, [2]=last_pred
    __shared__ _Float16 gsum[16 * 512];      // r,z gates: gi+gh+biases
    __shared__ _Float16 gni[16 * 256];       // n gate: gi + b_i
    __shared__ _Float16 gnh[16 * 256];       // n gate: gh + b_h

    const _Float16* Wp   = wf16;                    // 768x256
    const _Float16* Whh0 = wf16 + 1 * WROWS * HID;
    const _Float16* Wih1 = wf16 + 2 * WROWS * HID;
    const _Float16* Whh1 = wf16 + 3 * WROWS * HID;
    const _Float16* Wout = wf16 + 4 * WROWS * HID;  // 256x256

    const int t    = threadIdx.x;
    const int wave = t >> 5;                 // 0..15
    const int lane = t & 31;
    const int b0   = blockIdx.x << 4;
    const int pcol = t & 255;                // pointwise column owned by thread
    const int prow0 = (t >> 8);              // 0 or 1

    // ---- init: h0 = h1 = context, last_pred = 0 -----------------------------
#pragma unroll 1
    for (int i = 0; i < 8; ++i) {
        const int row = 2 * i + prow0;
        _Float16 c = ctx[(size_t)(b0 + row) * 256 + pcol];
        hbuf[0][row * HSTR + pcol] = c;
        hbuf[1][row * HSTR + pcol] = c;
        hbuf[2][row * HSTR + pcol] = (_Float16)0.0f;
    }
    __syncthreads();

    // gates: gi = aI @ Wi^T (+bI),  gh = aH @ Wh^T (+bH); r,z summed, n split.
    auto gates = [&](const _Float16* aI, const _Float16* aH,
                     const _Float16* Wi, const _Float16* Wh,
                     const float* __restrict__ bI, const float* __restrict__ bH) {
#pragma unroll 1
        for (int j = 0; j < 3; ++j) {                 // 16 waves x 3 = 48 col tiles
            const int n0 = (wave * 3 + j) << 4;
            f8 ai = {}; f8 ah = {};
#pragma unroll
            for (int k0 = 0; k0 < HID; k0 += 32)
                ai = wmma16x16x32(load_a_lds(aI, lane, k0),
                                  load_b_wt(Wi, lane, n0, k0), ai);
#pragma unroll
            for (int k0 = 0; k0 < HID; k0 += 32)
                ah = wmma16x16x32(load_a_lds(aH, lane, k0),
                                  load_b_wt(Wh, lane, n0, k0), ah);
            const int col   = n0 + (lane & 15);
            const float bi  = bI[col];
            const float bh  = bH[col];
            const int rbase = (lane >> 4) << 3;       // C/D layout: row = v + 8*(lane/16)
            if (n0 < 512) {                           // r and z gates
#pragma unroll
                for (int v = 0; v < 8; ++v)
                    gsum[(rbase + v) * 512 + col] = (_Float16)(ai[v] + ah[v] + bi + bh);
            } else {                                  // n gate: keep gi/gh separate
                const int c2 = col - 512;
#pragma unroll
                for (int v = 0; v < 8; ++v) {
                    gni[(rbase + v) * 256 + c2] = (_Float16)(ai[v] + bi);
                    gnh[(rbase + v) * 256 + c2] = (_Float16)(ah[v] + bh);
                }
            }
        }
    };

    // pointwise GRU update of hbuf[hidx]; thread owns column pcol, 8 rows.
    auto pointwise = [&](int hidx) {
#pragma unroll 1
        for (int i = 0; i < 8; ++i) {
            const int row = 2 * i + prow0;
            const float r = sigmoidf_((float)gsum[row * 512 + pcol]);
            const float z = sigmoidf_((float)gsum[row * 512 + 256 + pcol]);
            const float n = tanhf((float)gni[row * 256 + pcol] +
                                  r * (float)gnh[row * 256 + pcol]);
            const float hold = (float)hbuf[hidx][row * HSTR + pcol];
            hbuf[hidx][row * HSTR + pcol] = (_Float16)((1.0f - z) * n + z * hold);
        }
    };

    // ---- 12 autoregressive steps -------------------------------------------
#pragma unroll 1
    for (int s = 0; s < 12; ++s) {
        // layer 0: x = [step_emb (folded into embbias), last_pred]
        gates(hbuf[2], hbuf[0], Wp, Whh0, embbias + (size_t)s * WROWS, b_hh0);
        __syncthreads();
        pointwise(0);
        __syncthreads();
        // layer 1
        gates(hbuf[0], hbuf[1], Wih1, Whh1, b_ih1, b_hh1);
        __syncthreads();
        pointwise(1);
        __syncthreads();
        // prediction: pred = h1 @ Wout^T + b_out  (16 waves x 1 col tile)
        {
            const int n0 = wave << 4;
            f8 acc = {};
#pragma unroll
            for (int k0 = 0; k0 < HID; k0 += 32)
                acc = wmma16x16x32(load_a_lds(hbuf[1], lane, k0),
                                   load_b_wt(Wout, lane, n0, k0), acc);
            const int col   = n0 + (lane & 15);
            const float bo  = b_out[col];
            const int rbase = (lane >> 4) << 3;
#pragma unroll
            for (int v = 0; v < 8; ++v) {
                const int row = rbase + v;
                const float p = acc[v] + bo;
                out[((size_t)(b0 + row) * 12 + s) * 256 + col] = p;
                hbuf[2][row * HSTR + col] = (_Float16)p;
            }
        }
        __syncthreads();
    }
}

// ---------------------------------------------------------------------------
extern "C" void kernel_launch(void* const* d_in, const int* in_sizes, int n_in,
                              void* d_out, int out_size, void* d_ws, size_t ws_size,
                              hipStream_t stream) {
    (void)in_sizes; (void)n_in; (void)out_size; (void)ws_size;
    const float* ef   = (const float*)d_in[0];
    const float* semb = (const float*)d_in[1];
    const float* Wih0 = (const float*)d_in[2];
    const float* Whh0 = (const float*)d_in[3];
    const float* bih0 = (const float*)d_in[4];
    const float* bhh0 = (const float*)d_in[5];
    const float* Wih1 = (const float*)d_in[6];
    const float* Whh1 = (const float*)d_in[7];
    const float* bih1 = (const float*)d_in[8];
    const float* bhh1 = (const float*)d_in[9];
    const float* Wout = (const float*)d_in[10];
    const float* bout = (const float*)d_in[11];
    // d_in[12] = num_steps (== 12, fixed by output shape)

    // workspace layout (bytes):
    //   [0, 1703936)            f16 weights: Wp|Whh0|Wih1|Whh1|Wout
    //   [1703936, 1740800)      f32 embbias 12x768
    //   [1740800, 2265088)      f16 context 1024x256
    char* ws = (char*)d_ws;
    _Float16* wf16    = (_Float16*)ws;
    float*    embbias = (float*)(ws + 1703936);
    _Float16* ctx     = (_Float16*)(ws + 1703936 + 36864);

    const int cvt_total = 4 * 768 * 256 + 256 * 256;  // 851,968
    cvt_weights_kernel<<<(cvt_total + 255) / 256, 256, 0, stream>>>(
        Wih0, Whh0, Wih1, Whh1, Wout, wf16);
    emb_bias_kernel<<<(12 * 768 + 255) / 256, 256, 0, stream>>>(
        semb, Wih0, bih0, embbias);
    context_kernel<<<1024, 256, 0, stream>>>(ef, ctx);
    ar_gru_kernel<<<1024 / 16, NTHREADS, 0, stream>>>(
        wf16, embbias, ctx, bhh0, bih1, bhh1, bout, (float*)d_out);
}
